// PositionwiseFeedForward_2705829397358
// MI455X (gfx1250) — 8-bit matrix operand order repaired and checked by reading, compiled, not run on hardware
//
#include <hip/hip_runtime.h>


typedef __attribute__((ext_vector_type(16))) int          v16i;
typedef __attribute__((ext_vector_type(8)))  float        v8f;
typedef __attribute__((ext_vector_type(4)))  unsigned int tdm_g0_t;
typedef __attribute__((ext_vector_type(8)))  int          tdm_g1_t;
typedef __attribute__((ext_vector_type(4)))  int          tdm_g2_t;

#define BM 128
#define BN 128
#define BK 128
#define THREADS 256

__device__ __forceinline__ int clamp_i(int v, int lo, int hi) {
    return v < lo ? lo : (v > hi ? hi : v);
}

__device__ __forceinline__ unsigned fp8_mag(int n) {
    const unsigned long long lo = 0x4E4C4A4844403800ULL;
    const unsigned long long hi = 0x5756555453525150ULL;
    unsigned long long tbl = (n & 8) ? hi : lo;
    return (unsigned)((tbl >> ((n & 7) * 8)) & 0xFFu);
}
__device__ __forceinline__ unsigned fp8_sint(int q) {
    int n = q < 0 ? -q : q;
    return fp8_mag(n) | (q < 0 ? 0x80u : 0x00u);
}

__device__ __forceinline__ void tdm_load_tile_2d(const void* gtile, unsigned lds_off,
                                                 unsigned row_bytes,
                                                 unsigned total_rows,
                                                 unsigned tile_bytes,
                                                 unsigned tile_rows) {
    const unsigned d0_units     = row_bytes >> 3;
    const unsigned stride_units = row_bytes >> 3;
    const unsigned t0_units     = tile_bytes >> 3;

    unsigned long long ga = (unsigned long long)(uintptr_t)gtile;
    tdm_g0_t g0;
    g0.x = 1u;
    g0.y = lds_off;
    g0.z = (unsigned)ga;
    g0.w = ((unsigned)(ga >> 32) & 0x01FFFFFFu)
           | (2u << 30);
    tdm_g1_t g1;
    g1[0] = (int)(3u << 16);
    g1[1] = (int)((d0_units & 0xFFFFu) << 16);
    g1[2] = (int)(((d0_units >> 16) & 0xFFFFu)
           | ((total_rows & 0xFFFFu) << 16));
    g1[3] = (int)(((total_rows >> 16) & 0xFFFFu)
           | ((t0_units & 0xFFFFu) << 16));
    g1[4] = (int)(tile_rows & 0xFFFFu);
    g1[5] = (int)stride_units;
    g1[6] = 0;
    g1[7] = 0;
    tdm_g2_t z4 = {0, 0, 0, 0};
    tdm_g1_t z8 = {0, 0, 0, 0, 0, 0, 0, 0};
    __builtin_amdgcn_tensor_load_to_lds(g0, g1, z4, z4, z8, 0);
}

__global__ void ffn_init_scales(float* scales) {
    if (threadIdx.x < 2) scales[threadIdx.x] = 0.0f;
}

__global__ __launch_bounds__(THREADS)
void ffn_maxabs(const float* __restrict__ x, int n4, float* __restrict__ out) {
    float m = 0.0f;
    int stride = gridDim.x * blockDim.x;
    for (int i = blockIdx.x * blockDim.x + threadIdx.x; i < n4; i += stride) {
        float4 v = ((const float4*)x)[i];
        m = fmaxf(m, fmaxf(fmaxf(fabsf(v.x), fabsf(v.y)), fmaxf(fabsf(v.z), fabsf(v.w))));
    }
    for (int off = 16; off > 0; off >>= 1)
        m = fmaxf(m, __shfl_xor(m, off, 32));
    __shared__ float sm[8];
    int lane = threadIdx.x & 31, wid = threadIdx.x >> 5;
    if (lane == 0) sm[wid] = m;
    __syncthreads();
    if (threadIdx.x < 8) {
        m = sm[threadIdx.x];
        for (int off = 4; off > 0; off >>= 1)
            m = fmaxf(m, __shfl_xor(m, off, 32));
        if (threadIdx.x == 0)
            atomicMax((unsigned int*)out, __float_as_uint(m));
    }
}

template<int DIV>
__global__ __launch_bounds__(THREADS)
void ffn_quant_s4_fp8(const float* __restrict__ x, const float* __restrict__ scale_ptr,
                      unsigned char* __restrict__ q, int n4) {
    int i = blockIdx.x * blockDim.x + threadIdx.x;
    if (i >= n4) return;
    float inv = (float)DIV / scale_ptr[0];
    float4 v = ((const float4*)x)[i];
    int q0 = clamp_i((int)rintf(v.x * inv), -8, 7);
    int q1 = clamp_i((int)rintf(v.y * inv), -8, 7);
    int q2 = clamp_i((int)rintf(v.z * inv), -8, 7);
    int q3 = clamp_i((int)rintf(v.w * inv), -8, 7);
    ((unsigned int*)q)[i] = fp8_sint(q0) | (fp8_sint(q1) << 8) |
                            (fp8_sint(q2) << 16) | (fp8_sint(q3) << 24);
}

template<bool QUANT_OUT>
__global__ __launch_bounds__(THREADS)
void ffn_gemm_fp8_wmma(const unsigned char* __restrict__ A,
                       const unsigned char* __restrict__ Bm,
                       int M, int N, int K,
                       const float* __restrict__ bias,
                       const float* __restrict__ s_in_ptr,
                       const float* __restrict__ sw_maxabs_ptr,
                       const float* __restrict__ s_out_ptr,
                       unsigned char* __restrict__ qout,
                       float* __restrict__ fout)
{
    __shared__ __align__(16) unsigned char As[2][BM * BK];
    __shared__ __align__(16) unsigned char Bs[2][BN * BK];

    const int t    = threadIdx.x;
    const int lane = t & 31;
    const int wave = t >> 5;
    const int wm   = wave & 3;
    const int wn   = wave >> 2;
    const int mBase = blockIdx.y * BM;
    const int nBase = blockIdx.x * BN;

    const unsigned offA[2] = { (unsigned)(uintptr_t)&As[0][0], (unsigned)(uintptr_t)&As[1][0] };
    const unsigned offB[2] = { (unsigned)(uintptr_t)&Bs[0][0], (unsigned)(uintptr_t)&Bs[1][0] };
    const unsigned char* Atile = A  + (size_t)mBase * K;
    const unsigned char* Btile = Bm + (size_t)nBase * K;

    v8f zero = {};
    v8f acc[2][4];
#pragma unroll
    for (int i = 0; i < 2; ++i)
#pragma unroll
        for (int j = 0; j < 4; ++j) acc[i][j] = zero;

    const int hiA  = (lane >> 4) << 1;
    const int hiB  = (lane >> 4) << 2;
    const int rsel = lane & 15;

    const int nIter = K / BK;

    if (wave == 0) {
        tdm_load_tile_2d(Atile, offA[0], (unsigned)K, (unsigned)M, BK, BM);
        tdm_load_tile_2d(Btile, offB[0], (unsigned)K, (unsigned)N, BK, BN);
        __builtin_amdgcn_s_wait_tensorcnt(0);
    }
    __syncthreads();

    for (int ki = 0; ki < nIter; ++ki) {
        const int cur = ki & 1;
        const int nxt = cur ^ 1;

        if (wave == 0 && (ki + 1) < nIter) {
            const size_t k0n = (size_t)(ki + 1) * BK;
            tdm_load_tile_2d(Atile + k0n, offA[nxt], (unsigned)K, (unsigned)M, BK, BM);
            tdm_load_tile_2d(Btile + k0n, offB[nxt], (unsigned)K, (unsigned)N, BK, BN);
        }

        v16i af[2];
#pragma unroll
        for (int tm = 0; tm < 2; ++tm) {
            const int* ar = (const int*)(&As[cur][0] + (size_t)(wm * 32 + tm * 16 + rsel) * BK);
#pragma unroll
            for (int p = 0; p < 4; ++p) {
                int d = p * 4 + hiA;
                af[tm][2 * p]         = ar[d];
                af[tm][2 * p + 1]     = ar[d + 1];
                af[tm][8 + 2 * p]     = ar[d + 16];
                af[tm][8 + 2 * p + 1] = ar[d + 17];
            }
        }
#pragma unroll
        for (int tn = 0; tn < 4; ++tn) {
            const int* br = (const int*)(&Bs[cur][0] + (size_t)(wn * 64 + tn * 16 + rsel) * BK);
            v16i bf;
#pragma unroll
            for (int g = 0; g < 4; ++g)
#pragma unroll
                for (int j = 0; j < 4; ++j)
                    bf[4 * g + j] = br[g * 8 + (j >> 1) * 4 + (j & 1) + (hiB >> 1)];
#pragma unroll
            for (int tm = 0; tm < 2; ++tm)
                acc[tm][tn] = __builtin_amdgcn_wmma_f32_16x16x128_fp8_fp8(
                    af[tm], bf, (short)0, acc[tm][tn], false, false);
        }

        if (wave == 0) __builtin_amdgcn_s_wait_tensorcnt(0);
        __syncthreads();
    }

    const float s_in = s_in_ptr[0];
    const float sw   = sw_maxabs_ptr[0] * (1.0f / 7.0f);
    const float deq  = s_in * sw;
    float inv_s_out = 0.0f;
    if constexpr (QUANT_OUT) inv_s_out = 1.0f / s_out_ptr[0];

#pragma unroll
    for (int tn = 0; tn < 4; ++tn) {
        const int n = nBase + wn * 64 + tn * 16 + rsel;
        float bq = rintf(bias[n] / deq);
        bq = fminf(fmaxf(bq, -2.147483648e9f), 2.147483520e9f);
#pragma unroll
        for (int tm = 0; tm < 2; ++tm) {
            const int mRow = mBase + wm * 32 + tm * 16 + ((lane >> 4) << 3);
#pragma unroll
            for (int v = 0; v < 8; ++v) {
                const int   m  = mRow + v;
                const float hv = acc[tm][tn][v] + bq;
                if constexpr (QUANT_OUT) {
                    float hf = fmaxf(hv * deq, 0.0f);
                    int   qv = clamp_i((int)rintf(hf * inv_s_out), 0, 15);
                    qout[(size_t)m * N + n] = (unsigned char)fp8_mag(qv);
                } else {
                    fout[(size_t)m * N + n] = hv * deq;
                }
            }
        }
    }
}

extern "C" void kernel_launch(void* const* d_in, const int* in_sizes, int n_in,
                              void* d_out, int out_size, void* d_ws, size_t ws_size,
                              hipStream_t stream) {
    const float* x  = (const float*)d_in[0];
    const float* W1 = (const float*)d_in[1];
    const float* b1 = (const float*)d_in[2];
    const float* W2 = (const float*)d_in[3];
    const float* b2 = (const float*)d_in[4];
    const float* s1 = (const float*)d_in[5];
    const float* s2 = (const float*)d_in[6];

    const int F = in_sizes[2];
    const int D = in_sizes[4];
    const int M = in_sizes[0] / D;

    float*         scales = (float*)d_ws;
    unsigned char* qx  = (unsigned char*)d_ws + 256;
    unsigned char* qW1 = qx  + (size_t)M * D;
    unsigned char* qW2 = qW1 + (size_t)F * D;
    unsigned char* qh  = qW2 + (size_t)D * F;

    ffn_init_scales<<<1, 32, 0, stream>>>(scales);

    const int nW1_4 = (F * D) / 4;
    const int nW2_4 = (D * F) / 4;
    const int nX_4  = (M * D) / 4;
    int redBlk = 2048;
    ffn_maxabs<<<redBlk, THREADS, 0, stream>>>(W1, nW1_4, scales + 0);
    ffn_maxabs<<<redBlk, THREADS, 0, stream>>>(W2, nW2_4, scales + 1);

    ffn_quant_s4_fp8<1><<<(nX_4  + THREADS - 1) / THREADS, THREADS, 0, stream>>>(x,  s1,         qx,  nX_4);
    ffn_quant_s4_fp8<7><<<(nW1_4 + THREADS - 1) / THREADS, THREADS, 0, stream>>>(W1, scales + 0, qW1, nW1_4);
    ffn_quant_s4_fp8<7><<<(nW2_4 + THREADS - 1) / THREADS, THREADS, 0, stream>>>(W2, scales + 1, qW2, nW2_4);

    dim3 g1(F / BN, M / BM);
    ffn_gemm_fp8_wmma<true><<<g1, THREADS, 0, stream>>>(
        qx, qW1, M, F, D, b1, s1, scales + 0, s2, qh, nullptr);

    dim3 g2(D / BN, M / BM);
    ffn_gemm_fp8_wmma<false><<<g2, THREADS, 0, stream>>>(
        qh, qW2, M, D, F, b2, s2, scales + 1, nullptr, nullptr, (float*)d_out);
}
